// GraphSAGE_77086073028678
// MI455X (gfx1250) — matrix operand order repaired and checked by reading, not compiled, not run on hardware
//
#include <hip/hip_runtime.h>
#include <hip/hip_bf16.h>

// ---------------------------------------------------------------------------
// GraphSAGE on MI455X (gfx1250, wave32, WMMA).
//
//   memset(deg, msg1, msg2, h1)            (msg2/h1 are zero-padded to 32 ch)
//   pack 7 weight matrices -> f16 WMMA-B-layout tiles
//   deg[dst] += 1 ; msg1[dst] += x[src] ; msg1 /= clip(deg,1)
//   h1[:, :16] = relu(agg1@w1_n + x@w1_r + b1)          (WMMA <KT1=2,KT2=2>)
//   msg2[dst,:16] += h1[src,:16] ; msg2 /= clip(deg,1)
//   h2 = relu(agg2@w2_n + h1@w2_r + b2)                 (WMMA <1,1>, K padded)
//   h3 = relu(h2@fw1+fb1); h4 = relu(h3@fw2+fb2)        (WMMA <1,0>, <2,0>)
//   out = h4@fw3 + fb3                                  (WMMA <4,0>, N=2 pad 16)
//
// All K dims are multiples of 32 (conv2 zero-padded), so A loads are always
// unconditional 128-bit vector loads -> no divergent tail path.
// ---------------------------------------------------------------------------

typedef __attribute__((ext_vector_type(16))) _Float16 v16h;
typedef __attribute__((ext_vector_type(8)))  float    v8f;

static constexpr int kNodes = 50000;
static constexpr int kEdges = 800000;
static constexpr int kTM    = 4;      // m-tiles per wave (B reuse factor)

// ---------------------------------------------------------------------------
// Pack row-major f32 W[K,N] into f16 WMMA B-operand tiles.
// Tile (kt,nt) = 512 halves [lane][half]: lane = column (n = nt*16 + lane&15);
// lanes 0-15 hold K = kt*32+0..15 in halves 0..15, lanes 16-31 hold K 16..31.
// Out-of-range (K,N) -> 0 (free K/N padding).
// ---------------------------------------------------------------------------
__global__ void gsage_pack_b(const float* __restrict__ W, _Float16* __restrict__ out,
                             int K, int N, int NT, int total) {
  int idx = blockIdx.x * blockDim.x + threadIdx.x;
  if (idx >= total) return;
  int h    = idx & 15;
  int lane = (idx >> 4) & 31;
  int t    = idx >> 9;            // tile = kt*NT + nt
  int kt   = t / NT;
  int nt   = t - kt * NT;
  int n    = nt * 16 + (lane & 15);
  int k    = kt * 32 + (lane >> 4) * 16 + h;
  float v  = (k < K && n < N) ? W[(size_t)k * N + n] : 0.0f;
  out[idx] = (_Float16)v;
}

// deg[dst] += 1
__global__ void gsage_degree(const long long* __restrict__ dst,
                             float* __restrict__ deg, int E) {
  int e = blockIdx.x * blockDim.x + threadIdx.x;
  if (e < E) unsafeAtomicAdd(&deg[dst[e]], 1.0f);
}

// msg[dst*mStride + c] += X[src*xStride + c], c in [0, 1<<logD)
__global__ void gsage_scatter(const float* __restrict__ X,
                              const long long* __restrict__ src,
                              const long long* __restrict__ dst,
                              float* __restrict__ msg, int E, int logD,
                              int xStride, int mStride) {
  int idx = blockIdx.x * blockDim.x + threadIdx.x;
  int total = E << logD;
  if (idx >= total) return;
  int e = idx >> logD;
  int c = idx & ((1 << logD) - 1);
  long long s = src[e], d = dst[e];
  unsafeAtomicAdd(&msg[(size_t)d * mStride + c], X[(size_t)s * xStride + c]);
}

// msg[n, c] /= max(deg[n], 1)   over a contiguous [Nnodes << logD] buffer
__global__ void gsage_normalize(float* __restrict__ msg, const float* __restrict__ deg,
                                int Nnodes, int logD) {
  int idx = blockIdx.x * blockDim.x + threadIdx.x;
  int total = Nnodes << logD;
  if (idx >= total) return;
  float d = deg[idx >> logD];
  d = d < 1.0f ? 1.0f : d;
  msg[idx] = msg[idx] * (1.0f / d);
}

// ---------------------------------------------------------------------------
// Unconditional A-chunk load: 32 f32 of one row (16x32 tile, contiguous K per lane group, as packed B).
// p0 points at A[row*lda + kbase]; lane group g = lane>>4.
// halves 0..15 = K g*16..g*16+15.
// ---------------------------------------------------------------------------
__device__ __forceinline__ v16h gsage_load_a32(const float* __restrict__ p0, int g) {
  const float* p = p0 + g * 16;
  float4 f0 = ((const float4*)p)[0];
  float4 f1 = ((const float4*)p)[1];
  float4 f2 = ((const float4*)(p + 8))[0];
  float4 f3 = ((const float4*)(p + 8))[1];
  v16h a;
  a[0]=(_Float16)f0.x;  a[1]=(_Float16)f0.y;  a[2]=(_Float16)f0.z;  a[3]=(_Float16)f0.w;
  a[4]=(_Float16)f1.x;  a[5]=(_Float16)f1.y;  a[6]=(_Float16)f1.z;  a[7]=(_Float16)f1.w;
  a[8]=(_Float16)f2.x;  a[9]=(_Float16)f2.y;  a[10]=(_Float16)f2.z; a[11]=(_Float16)f2.w;
  a[12]=(_Float16)f3.x; a[13]=(_Float16)f3.y; a[14]=(_Float16)f3.z; a[15]=(_Float16)f3.w;
  return a;
}

// ---------------------------------------------------------------------------
// Fused GEMM: C = act( A1@B1 [+ A2@B2] + bias ). One wave computes kTM
// consecutive 16x16 output tiles in one column strip, with all B tiles held
// in registers (loaded once). KT1/KT2 = K chunks of 32, compile-time.
// ---------------------------------------------------------------------------
template <int KT1, int KT2, bool RELU>
__global__ void gsage_gemm(const float* __restrict__ A1,
                           const float* __restrict__ A2,
                           const _Float16* __restrict__ B1,
                           const _Float16* __restrict__ B2,
                           const float* __restrict__ bias,
                           float* __restrict__ C,
                           int MT, int Nout, int NT, int ldc) {
  int wid  = blockIdx.x * (blockDim.x >> 5) + (threadIdx.x >> 5);
  int lane = threadIdx.x & 31;
  int groups = (MT + kTM - 1) / kTM;
  if (wid >= groups * NT) return;          // uniform per wave (EXEC all-ones)
  int gidx = wid / NT;
  int nt   = wid - gidx * NT;
  int mt0  = gidx * kTM;
  int g    = lane >> 4;
  int col  = nt * 16 + (lane & 15);
  float bv = (col < Nout) ? bias[col] : 0.0f;

  // B operands: one contiguous 32B v16h per lane per K-chunk, kept in VGPRs.
  v16h b1r[KT1];
  #pragma unroll
  for (int kt = 0; kt < KT1; ++kt)
    b1r[kt] = *(const v16h*)(B1 + (((size_t)(kt * NT + nt) * 32) + lane) * 16);
  v16h b2r[KT2 > 0 ? KT2 : 1];
  if constexpr (KT2 > 0) {
    #pragma unroll
    for (int kt = 0; kt < KT2; ++kt)
      b2r[kt] = *(const v16h*)(B2 + (((size_t)(kt * NT + nt) * 32) + lane) * 16);
  }

  #pragma unroll
  for (int tm = 0; tm < kTM; ++tm) {
    int mt = mt0 + tm;
    if (mt >= MT) break;                   // uniform per wave
    int arow = mt * 16 + (lane & 15);
    v8f c = {};
    #pragma unroll
    for (int kt = 0; kt < KT1; ++kt) {
      v16h a = gsage_load_a32(A1 + (size_t)arow * (KT1 * 32) + kt * 32, g);
      c = __builtin_amdgcn_wmma_f32_16x16x32_f16(false, a, false, b1r[kt],
                                                 (short)0, c, false, false);
    }
    if constexpr (KT2 > 0) {
      #pragma unroll
      for (int kt = 0; kt < KT2; ++kt) {
        v16h a = gsage_load_a32(A2 + (size_t)arow * (KT2 * 32) + kt * 32, g);
        c = __builtin_amdgcn_wmma_f32_16x16x32_f16(false, a, false, b2r[kt],
                                                   (short)0, c, false, false);
      }
    }
    // C/D layout: lane group g -> rows mt*16 + g*8 + r, column = col.
    if (col < Nout) {
      #pragma unroll
      for (int r = 0; r < 8; ++r) {
        float v = c[r] + bv;
        if constexpr (RELU) v = v > 0.0f ? v : 0.0f;
        C[(size_t)(mt * 16 + g * 8 + r) * ldc + col] = v;
      }
    }
  }
}

// ---------------------------------------------------------------------------
// Host launcher
// ---------------------------------------------------------------------------
extern "C" void kernel_launch(void* const* d_in, const int* in_sizes, int n_in,
                              void* d_out, int out_size, void* d_ws, size_t ws_size,
                              hipStream_t stream) {
  const float*     x  = (const float*)d_in[0];
  const long long* ei = (const long long*)d_in[1];   // int64 [2, E]
  const float* w1n = (const float*)d_in[2];
  const float* w1r = (const float*)d_in[3];
  const float* b1  = (const float*)d_in[4];
  const float* w2n = (const float*)d_in[5];
  const float* w2r = (const float*)d_in[6];
  const float* b2  = (const float*)d_in[7];
  const float* fw1 = (const float*)d_in[8];
  const float* fb1 = (const float*)d_in[9];
  const float* fw2 = (const float*)d_in[10];
  const float* fb2 = (const float*)d_in[11];
  const float* fw3 = (const float*)d_in[12];
  const float* fb3 = (const float*)d_in[13];
  float* out = (float*)d_out;

  const long long* srcI = ei;
  const long long* dstI = ei + kEdges;

  // ---- workspace layout (256B aligned) ----
  char* base = (char*)d_ws;
  size_t cur = 0;
  auto alloc = [&](size_t bytes) -> char* {
    char* p = base + cur;
    cur = (cur + bytes + 255) & ~(size_t)255;
    return p;
  };
  float* deg  = (float*)alloc((size_t)kNodes * 4);         // [N]
  float* msg1 = (float*)alloc((size_t)kNodes * 64 * 4);    // [N,64] -> agg1
  float* h1   = (float*)alloc((size_t)kNodes * 32 * 4);    // [N,32] cols 16..31 = 0
  float* msg2 = (float*)alloc((size_t)kNodes * 32 * 4);    // [N,32] cols 16..31 = 0
  float* h2   = (float*)alloc((size_t)kNodes * 32 * 4);    // [N,32]
  float* h3   = (float*)alloc((size_t)kNodes * 64 * 4);    // [N,64]
  float* h4   = (float*)alloc((size_t)kNodes * 128 * 4);   // [N,128]
  _Float16* pw1n = (_Float16*)alloc(2 * 1 * 512 * 2);      // K=64  N=16
  _Float16* pw1r = (_Float16*)alloc(2 * 1 * 512 * 2);      // K=64  N=16
  _Float16* pw2n = (_Float16*)alloc(1 * 2 * 512 * 2);      // K=16(pad32) N=32
  _Float16* pw2r = (_Float16*)alloc(1 * 2 * 512 * 2);      // K=16(pad32) N=32
  _Float16* pfw1 = (_Float16*)alloc(1 * 4 * 512 * 2);      // K=32  N=64
  _Float16* pfw2 = (_Float16*)alloc(2 * 8 * 512 * 2);      // K=64  N=128
  _Float16* pfw3 = (_Float16*)alloc(4 * 1 * 512 * 2);      // K=128 N=2(pad16)

  // ---- zero accumulation + padded buffers (capture-safe memset nodes) ----
  hipMemsetAsync(deg,  0, (size_t)kNodes * 4,      stream);
  hipMemsetAsync(msg1, 0, (size_t)kNodes * 64 * 4, stream);
  hipMemsetAsync(h1,   0, (size_t)kNodes * 32 * 4, stream);
  hipMemsetAsync(msg2, 0, (size_t)kNodes * 32 * 4, stream);

  // ---- pack weights ----
  auto pack = [&](const float* W, _Float16* P, int K, int N, int KT, int NT) {
    int total = KT * NT * 512;
    gsage_pack_b<<<(total + 255) / 256, 256, 0, stream>>>(W, P, K, N, NT, total);
  };
  pack(w1n, pw1n, 64, 16, 2, 1);
  pack(w1r, pw1r, 64, 16, 2, 1);
  pack(w2n, pw2n, 16, 32, 1, 2);
  pack(w2r, pw2r, 16, 32, 1, 2);
  pack(fw1, pfw1, 32, 64, 1, 4);
  pack(fw2, pfw2, 64, 128, 2, 8);
  pack(fw3, pfw3, 128, 2, 4, 1);

  // ---- degrees ----
  gsage_degree<<<(kEdges + 255) / 256, 256, 0, stream>>>(dstI, deg, kEdges);

  const int MT = kNodes >> 4;                 // 3125 m-tiles
  auto gemmGrid = [&](int NT) {
    int waves = ((MT + kTM - 1) / kTM) * NT;
    return (waves + 7) / 8;                   // 8 waves / 256-thread block
  };

  // ---- conv1: aggregate x (64ch); h1[:, :16] = relu(agg1@w1n + x@w1r + b1) ----
  {
    int total = kEdges << 6;
    gsage_scatter<<<(total + 255) / 256, 256, 0, stream>>>(
        x, srcI, dstI, msg1, kEdges, 6, 64, 64);
    int nt = kNodes << 6;
    gsage_normalize<<<(nt + 255) / 256, 256, 0, stream>>>(msg1, deg, kNodes, 6);
  }
  gsage_gemm<2, 2, true><<<gemmGrid(1), 256, 0, stream>>>(
      msg1, x, pw1n, pw1r, b1, h1, MT, /*Nout=*/16, /*NT=*/1, /*ldc=*/32);

  // ---- conv2: aggregate h1 (16ch, stride 32); h2 = relu(...) ----
  {
    int total = kEdges << 4;
    gsage_scatter<<<(total + 255) / 256, 256, 0, stream>>>(
        h1, srcI, dstI, msg2, kEdges, 4, 32, 32);
    int nt = kNodes << 5;   // normalize padded [N,32]; zeros stay zero
    gsage_normalize<<<(nt + 255) / 256, 256, 0, stream>>>(msg2, deg, kNodes, 5);
  }
  gsage_gemm<1, 1, true><<<gemmGrid(2), 256, 0, stream>>>(
      msg2, h1, pw2n, pw2r, b2, h2, MT, 32, 2, 32);

  // ---- MLP head ----
  gsage_gemm<1, 0, true><<<gemmGrid(4), 256, 0, stream>>>(
      h2, nullptr, pfw1, nullptr, fb1, h3, MT, 64, 4, 64);
  gsage_gemm<2, 0, true><<<gemmGrid(8), 256, 0, stream>>>(
      h3, nullptr, pfw2, nullptr, fb2, h4, MT, 128, 8, 128);
  gsage_gemm<4, 0, false><<<gemmGrid(1), 256, 0, stream>>>(
      h4, nullptr, pfw3, nullptr, fb3, out, MT, 2, 1, 2);
}
